// QKRotationWrapper_21887153340654
// MI455X (gfx1250) — matrix operand order repaired and checked by reading, not compiled, not run on hardware
//
#include <hip/hip_runtime.h>
#include <hip/hip_bf16.h>
#include <stdint.h>

// ext-vector types matching the gfx1250 WMMA builtin signatures
typedef __attribute__((ext_vector_type(16))) __bf16         v16bf;
typedef __attribute__((ext_vector_type(8)))  float          v8f;
typedef __attribute__((ext_vector_type(4)))  float          v4f;   // true clang vector (NT-load OK)
typedef __attribute__((ext_vector_type(16))) unsigned short v16u16;

#define HEAD 128
#define HAD_SCALE 0.08838834764831845f   // 1/sqrt(128)

__device__ __forceinline__ unsigned short bf16_rne(float f) {
  uint32_t u = __builtin_bit_cast(uint32_t, f);
  u += 0x7FFFu + ((u >> 16) & 1u);        // round-to-nearest-even
  return (unsigned short)(u >> 16);
}
__device__ __forceinline__ float bf16_up(unsigned short h) {
  return __builtin_bit_cast(float, (uint32_t)h << 16);
}

// One wave32 handles 16x128 row tiles:  D(16x128) = X(16x128) * H(128x128) * scale
// H built once per block in LDS as bf16 (+-1 exact); grid-stride tile loop
// amortizes the fill. X is split x = hi + lo (both bf16) so the f32-accumulated
// bf16 WMMA result carries ~16 mantissa bits (near-f32 accuracy; products by +-1
// are exact). Streaming loads/stores use non-temporal hints: each element is
// touched exactly once, so don't let 671 MB of stream claim L2 residency.
__global__ void __launch_bounds__(256)
had128_wmma_kernel(const float* __restrict__ in, float* __restrict__ out, int ntiles) {
  __shared__ unsigned short Hs[HEAD * HEAD];   // 32 KB bf16 Hadamard matrix
  const int tid = threadIdx.x;

  // Build H in LDS: H[k][n] = (-1)^popcount(k&n), packed 2 bf16 per dword.
  unsigned int* H32 = reinterpret_cast<unsigned int*>(Hs);
  #pragma unroll
  for (int i = 0; i < (HEAD * HEAD / 2) / 256; ++i) {
    int idx  = i * 256 + tid;
    int krow = idx >> 6;              // 64 dwords (128 bf16) per row
    int n0   = (idx & 63) * 2;
    unsigned int lo = (__builtin_popcount(krow & n0)       & 1) ? 0xBF80u : 0x3F80u;
    unsigned int hi = (__builtin_popcount(krow & (n0 + 1)) & 1) ? 0xBF80u : 0x3F80u;
    H32[idx] = (hi << 16) | lo;
  }
  __syncthreads();

  const int wave   = tid >> 5;        // 8 waves per block
  const int lane   = tid & 31;
  const int laneHi = lane >> 4;       // 0 for lanes 0-15, 1 for lanes 16-31
  const int lm     = lane & 15;
  const int kb     = lm + laneHi * 16;   // B-fragment K within a 32-chunk

  // Grid-stride over 16-row tiles; wave-uniform loop keeps EXEC all-ones.
  for (int t = blockIdx.x * 8 + wave; t < ntiles; t += gridDim.x * 8) {
    const int row0 = t * 16;

    // ---- Load A tile (16 rows x 128) as 4 K-chunks of 32, split hi/lo bf16 ----
    // A k order used here (16x32, contiguous, B matches): lanes 0-15 hold M=0..15, K = 0..15;
    // lanes 16-31 hold M=0..15, K = 16..31. Packed pairs per VGPR.
    v16bf a_hi[4], a_lo[4];
    const float* rowp = in + (size_t)(row0 + lm) * HEAD + laneHi * 16;
    #pragma unroll
    for (int kc = 0; kc < 4; ++kc) {
      const float* p = rowp + kc * 32;
      v4f x0 = __builtin_nontemporal_load((const v4f*)(p + 0));
      v4f x1 = __builtin_nontemporal_load((const v4f*)(p + 4));
      v4f x2 = __builtin_nontemporal_load((const v4f*)(p + 8));
      v4f x3 = __builtin_nontemporal_load((const v4f*)(p + 12));
      float x[16] = {x0.x, x0.y, x0.z, x0.w, x1.x, x1.y, x1.z, x1.w,
                     x2.x, x2.y, x2.z, x2.w, x3.x, x3.y, x3.z, x3.w};
      v16u16 h, l;
      #pragma unroll
      for (int j = 0; j < 16; ++j) {
        unsigned short bh = bf16_rne(x[j]);
        h[j] = bh;
        l[j] = bf16_rne(x[j] - bf16_up(bh));   // exact residual split
      }
      a_hi[kc] = __builtin_bit_cast(v16bf, h);
      a_lo[kc] = __builtin_bit_cast(v16bf, l);
    }

    // B k order used here (32x16, same as A): lane holds N=lane%16, lanes 0-15 hold K=0..15,
    // lanes 16-31 hold K=16..31 -> H symmetric, lane reads 16 contiguous bf16 of row N.
    float* orow = out + (size_t)row0 * HEAD;

    #pragma unroll 2
    for (int nb = 0; nb < 8; ++nb) {
      v8f acc = {};
      #pragma unroll
      for (int kc = 0; kc < 4; ++kc) {
        // byte offset = ((nb*16+lm)*128 + kc*32 + laneHi*16)*2 : multiple of 32 -> b128 aligned
        const v16bf bfrag =
            *reinterpret_cast<const v16bf*>(Hs + (nb * 16 + lm) * HEAD + kc * 32 + laneHi * 16);
        acc = __builtin_amdgcn_wmma_f32_16x16x32_bf16(
            false, a_hi[kc], false, bfrag, (short)0, acc, false, false);
        acc = __builtin_amdgcn_wmma_f32_16x16x32_bf16(
            false, a_lo[kc], false, bfrag, (short)0, acc, false, false);
      }
      // C/D layout: VGPR r, lanes 0-15 -> M=r, N=lane; lanes 16-31 -> M=r+8
      const int col = nb * 16 + lm;
      #pragma unroll
      for (int r = 0; r < 8; ++r) {
        __builtin_nontemporal_store(acc[r] * HAD_SCALE,
                                    orow + (size_t)(r + laneHi * 8) * HEAD + col);
      }
    }
  }
}

extern "C" void kernel_launch(void* const* d_in, const int* in_sizes, int n_in,
                              void* d_out, int out_size, void* d_ws, size_t ws_size,
                              hipStream_t stream) {
  const float* q = (const float*)d_in[0];
  const float* k = (const float*)d_in[1];
  float* out = (float*)d_out;

  const size_t qsz = (size_t)in_sizes[0];   // 4*32*4096*128
  const size_t ksz = (size_t)in_sizes[1];   // 4*8*4096*128
  const int qtiles = (int)(qsz / (HEAD * 16));   // 32768 16-row tiles
  const int ktiles = (int)(ksz / (HEAD * 16));   // 8192

  dim3 block(256);  // 8 wave32s per block; grid-stride over tiles
  had128_wmma_kernel<<<dim3(1024), block, 0, stream>>>(q, out, qtiles);
  had128_wmma_kernel<<<dim3(256),  block, 0, stream>>>(k, out + qsz, ktiles);
}
